// fA_P4TransitionBlock2D_78572131713151
// MI455X (gfx1250) — matrix operand order repaired and checked by reading, not compiled, not run on hardware
//
#include <hip/hip_runtime.h>
#include <cstdint>
#include <cstddef>

// ---------------------------------------------------------------------------
// p4 attention-gated transition block for MI455X (gfx1250, wave32, WMMA).
// Heavy math = 32 GEMMs of [64x512]x[512x9216] run on v_wmma_f32_16x16x32_f16.
// ---------------------------------------------------------------------------

typedef __attribute__((ext_vector_type(16))) _Float16 v16h;
typedef __attribute__((ext_vector_type(8)))  _Float16 v8h;
typedef __attribute__((ext_vector_type(8)))  float    v8f;

#define NH     4
#define CIN    128
#define COUT   64
#define HH     96
#define WW     96
#define HW     9216          // 96*96
#define KIH    512           // CIN*NH   (GEMM K)
#define BATCH  8
#define NBR    32            // BATCH*NH
#define EPSBN  2e-5f

// ---- workspace byte offsets -------------------------------------------------
#define OFF_MEAN  ((size_t)0)                         // 128 f32
#define OFF_RSTD  ((size_t)512)                       // 128 f32
#define OFF_SSUM  ((size_t)1024)                      // 8*512 f32 (atomic add)
#define OFF_SMAX  ((size_t)17408)                     // 8*512 f32-bits (atomic max, u>=0)
#define OFF_FAVG  ((size_t)33792)                     // 8*4*9216 f32
#define OFF_FMAX  ((size_t)1213440)                   // 8*4*9216 f32
#define OFF_ACH   ((size_t)2393088)                   // 32*128*4 f32
#define OFF_ASP   ((size_t)2458624)                   // 32*4*9216 f32
#define OFF_WEFF  ((size_t)4817920)                   // 32*64*512 f16
#define OFF_UT    ((size_t)6915072)                   // 8*9216*512 f16  [b][yx][ih]
#define OFF_YWS   ((size_t)82412544)                  // 32*64*9216 f32

// ---------------------------------------------------------------------------
__global__ void k_zero(float* p, int n) {
  int i = blockIdx.x * blockDim.x + threadIdx.x;
  if (i < n) p[i] = 0.f;
}

// BatchNorm statistics: one block per channel, reduce over B*NH*HW = 294912.
__global__ void k_bnstats(const float* __restrict__ x,
                          float* __restrict__ mean, float* __restrict__ rstd) {
  int c = blockIdx.x, t = threadIdx.x;
  float s = 0.f, s2 = 0.f;
  for (int b = 0; b < BATCH; ++b) {
    const float* xp = x + ((size_t)(b * CIN + c)) * (NH * HW);
    for (int e = t; e < NH * HW; e += 256) {
      float v = xp[e];
      s += v; s2 += v * v;
    }
  }
  __shared__ float ls[256], ls2[256];
  ls[t] = s; ls2[t] = s2;
  __syncthreads();
  for (int st = 128; st > 0; st >>= 1) {
    if (t < st) { ls[t] += ls[t + st]; ls2[t] += ls2[t + st]; }
    __syncthreads();
  }
  if (t == 0) {
    const float inv = 1.f / (float)(BATCH * NH * HW);
    float m = ls[0] * inv;
    float v = ls2[0] * inv - m * m;
    mean[c] = m;
    rstd[c] = rsqrtf(v + EPSBN);
  }
}

// Normalize+ReLU, write u as f16 transposed [b][yx][ih]; channel mean/max per
// (b,h,yx) via LDS; spatial mean/max per (b,i,h) via global atomics (u >= 0).
__global__ void k_norm(const float* __restrict__ x,
                       const float* __restrict__ gamma, const float* __restrict__ beta,
                       const float* __restrict__ mean,  const float* __restrict__ rstd,
                       _Float16* __restrict__ ut,
                       float* __restrict__ favg, float* __restrict__ fmaxb,
                       float* __restrict__ ssum, unsigned* __restrict__ smax) {
  __shared__ _Float16 uarr[KIH * 32];                 // 32 KB
  const int nb  = HW / 32;                            // 288 tiles per batch
  const int b   = blockIdx.x / nb;
  const int yx0 = (blockIdx.x % nb) * 32;
  const int t   = threadIdx.x;

  for (int rep = 0; rep < 2; ++rep) {
    int ih = rep * 256 + t;
    int i = ih >> 2, h = ih & 3;
    float sc = gamma[i] * rstd[i];
    float sh = beta[i] - mean[i] * sc;
    const float* xp = x + ((size_t)((b * CIN + i) * NH + h)) * HW + yx0;
    _Float16* up = ut + ((size_t)b * HW + yx0) * KIH + ih;
    float lsum = 0.f, lmax = 0.f;
#pragma unroll
    for (int q = 0; q < 8; ++q) {
      float4 v = *(const float4*)(xp + q * 4);
      float u0 = fmaxf(fmaf(v.x, sc, sh), 0.f);
      float u1 = fmaxf(fmaf(v.y, sc, sh), 0.f);
      float u2 = fmaxf(fmaf(v.z, sc, sh), 0.f);
      float u3 = fmaxf(fmaf(v.w, sc, sh), 0.f);
      int yy = q * 4;
      uarr[ih * 32 + yy + 0] = (_Float16)u0;
      uarr[ih * 32 + yy + 1] = (_Float16)u1;
      uarr[ih * 32 + yy + 2] = (_Float16)u2;
      uarr[ih * 32 + yy + 3] = (_Float16)u3;
      up[(size_t)(yy + 0) * KIH] = (_Float16)u0;
      up[(size_t)(yy + 1) * KIH] = (_Float16)u1;
      up[(size_t)(yy + 2) * KIH] = (_Float16)u2;
      up[(size_t)(yy + 3) * KIH] = (_Float16)u3;
      lsum += (u0 + u1) + (u2 + u3);
      lmax = fmaxf(lmax, fmaxf(fmaxf(u0, u1), fmaxf(u2, u3)));
    }
    atomicAdd(&ssum[b * KIH + ih], lsum);
    atomicMax(&smax[b * KIH + ih], __float_as_uint(lmax));   // u >= 0
  }
  __syncthreads();
  if (t < 128) {                                     // reduce over i for (h,yy)
    int h = t >> 5, yy = t & 31;
    float fs = 0.f, fm = 0.f;
    for (int i = 0; i < CIN; ++i) {
      float v = (float)uarr[(i * 4 + h) * 32 + yy];
      fs += v; fm = fmaxf(fm, v);
    }
    size_t o = ((size_t)(b * NH + h)) * HW + yx0 + yy;
    favg[o]  = fs * (1.f / CIN);
    fmaxb[o] = fm;
  }
}

// Channel-attention MLP (mid=2), one block per (b,r,h), 128 threads over i.
__global__ void k_mlp(const float* __restrict__ ssum, const float* __restrict__ smaxf,
                      const float* __restrict__ w1, const float* __restrict__ w2,
                      float* __restrict__ ach) {
  int blk = blockIdx.x;                // b*16 + r*4 + h
  int b = blk >> 4, r = (blk >> 2) & 3, h = blk & 3;
  int i = threadIdx.x;
  int hp = (h - r) & 3;
  float sa = ssum[b * KIH + i * 4 + h] * (1.f / HW);
  float sm = smaxf[b * KIH + i * 4 + h];
  __shared__ float red[4][128];
  float w1a = w1[(0 * CIN + i) * 4 + hp];
  float w1b = w1[(1 * CIN + i) * 4 + hp];
  red[0][i] = w1a * sa; red[1][i] = w1b * sa;
  red[2][i] = w1a * sm; red[3][i] = w1b * sm;
  __syncthreads();
  for (int st = 64; st > 0; st >>= 1) {
    if (i < st) {
      red[0][i] += red[0][i + st]; red[1][i] += red[1][i + st];
      red[2][i] += red[2][i + st]; red[3][i] += red[3][i + st];
    }
    __syncthreads();
  }
  float h0 = fmaxf(red[0][0], 0.f) + fmaxf(red[2][0], 0.f);
  float h1 = fmaxf(red[1][0], 0.f) + fmaxf(red[3][0], 0.f);
  float o = w2[(i * 2 + 0) * 4 + hp] * h0 + w2[(i * 2 + 1) * 4 + hp] * h1;
  ach[((b * 4 + r) * CIN + i) * 4 + h] = 1.f / (1.f + __expf(-o));
}

// Fold a_ch into the rolled 1x1 kernel: Weff[br][o][ih] (f16).
__global__ void k_fold(const float* __restrict__ w, const float* __restrict__ ach,
                       _Float16* __restrict__ weff) {
  int e = blockIdx.x * 256 + threadIdx.x;            // < 32*64*512
  int br = e >> 15, rem = e & 32767;
  int o = rem >> 9, ihh = rem & 511;
  int i = ihh >> 2, h = ihh & 3;
  int r = br & 3;
  int hp = (h - r) & 3;
  float v = w[(o * CIN + i) * 4 + hp] * ach[(br * CIN + i) * 4 + h];
  weff[e] = (_Float16)v;
}

// Spatial attention: rotated/rolled 7x7 grouped conv over (avg,max) channel pools.
__global__ void k_spat(const float* __restrict__ favg, const float* __restrict__ fmaxb,
                       const float* __restrict__ wsp, float* __restrict__ asp) {
  __shared__ float wl[98];
  int blk = blockIdx.x;                // b*16 + r*4 + h
  int b = blk >> 4, r = (blk >> 2) & 3, h = blk & 3;
  int t = threadIdx.x;
  int hp = (h - r) & 3;
  if (t < 98) {
    int c = t / 49, kk = t % 49;
    int ky = kk / 7, kx = kk % 7;
    int sy, sx;
    switch (r) {                       // inverse index map of rot90^r (CCW)
      case 0:  sy = ky;     sx = kx;     break;
      case 1:  sy = kx;     sx = 6 - ky; break;
      case 2:  sy = 6 - ky; sx = 6 - kx; break;
      default: sy = 6 - kx; sx = ky;    break;
    }
    wl[t] = wsp[((c * NH + hp) * 7 + sy) * 7 + sx];
  }
  __syncthreads();
  const float* fa = favg  + ((size_t)(b * NH + h)) * HW;
  const float* fm = fmaxb + ((size_t)(b * NH + h)) * HW;
  for (int p = t; p < HW; p += 256) {
    int y = p / WW, xx0 = p % WW;
    float acc = 0.f;
    for (int ky = 0; ky < 7; ++ky) {
      int yy = y + ky - 3;
      if ((unsigned)yy >= (unsigned)HH) continue;
      const float* fay = fa + yy * WW;
      const float* fmy = fm + yy * WW;
      for (int kx = 0; kx < 7; ++kx) {
        int xx = xx0 + kx - 3;
        if ((unsigned)xx >= (unsigned)WW) continue;
        acc = fmaf(wl[ky * 7 + kx],      fay[xx], acc);
        acc = fmaf(wl[49 + ky * 7 + kx], fmy[xx], acc);
      }
    }
    asp[(size_t)blk * HW + p] = 1.f / (1.f + __expf(-acc));
  }
}

// Main GEMM per (b,r): [64 x 512] x [512 x 9216] with a_sp gating of the B
// operand, via v_wmma_f32_16x16x32_f16. 8 waves/block, 16 N-cols per wave,
// full M=64 per wave, Weff staged in 64KB LDS with XOR-swizzled 16B chunks.
__global__ void __launch_bounds__(256) k_gemm(const _Float16* __restrict__ weff,
                                              const _Float16* __restrict__ ut,
                                              const float* __restrict__ asp,
                                              float* __restrict__ yws) {
  __shared__ _Float16 aW[COUT * KIH];                 // 65536 B
  const int t  = threadIdx.x;
  const int br = blockIdx.y;
  const int b  = br >> 2;

  const _Float16* wg = weff + (size_t)br * COUT * KIH;
#pragma unroll
  for (int it = 0; it < 16; ++it) {                  // 4096 16B chunks
    int g = it * 256 + t;
    int row = g >> 6, c = g & 63;
    v8h v = *(const v8h*)(wg + g * 8);
    *(v8h*)(aW + row * KIH + (((c ^ (row & 7))) << 3)) = v;
  }
  __syncthreads();

  const int wave = t >> 5, lane = t & 31;
  const int half = lane >> 4;
  const int n = blockIdx.x * 128 + wave * 16 + (lane & 15);

  // per-lane a_sp gates: B-fragment element e has h = e & 3 (k = base+e, base%4==0)
  const float* ap = asp + (size_t)br * NH * HW + n;
  v16h aspv;
#pragma unroll
  for (int h = 0; h < 4; ++h) {
    _Float16 a = (_Float16)ap[(size_t)h * HW];
    aspv[h] = a; aspv[h + 4] = a; aspv[h + 8] = a; aspv[h + 12] = a;
  }

  const _Float16* ub = ut + ((size_t)b * HW + n) * KIH + half * 16;
  v8f acc[4];
#pragma unroll
  for (int mo = 0; mo < 4; ++mo) acc[mo] = (v8f){};

  for (int kt = 0; kt < 16; ++kt) {
    // B fragment (32x16 KxN): lane holds n, K = half*16 + [0..15] (contiguous)
    const _Float16* up = ub + kt * 32;
    v8h b0 = *(const v8h*)up;
    v8h b1 = *(const v8h*)(up + 8);
    v16h bf = __builtin_shufflevector(b0, b1, 0, 1, 2, 3, 4, 5, 6, 7,
                                      8, 9, 10, 11, 12, 13, 14, 15);
    bf = bf * aspv;                                  // packed f16 gating

    int c1 = kt * 4 + half * 2;                      // A chunk: k = kt*32 + half*16
#pragma unroll
    for (int mo = 0; mo < 4; ++mo) {
      int row = mo * 16 + (lane & 15);
      int sw = row & 7;
      const _Float16* lp = aW + row * KIH;
      v8h a0 = *(const v8h*)(lp + ((c1 ^ sw) << 3));         // k .. k+7
      v8h a1 = *(const v8h*)(lp + (((c1 + 1) ^ sw) << 3));   // k+8 .. k+15
      v16h af = __builtin_shufflevector(a0, a1, 0, 1, 2, 3, 4, 5, 6, 7,
                                        8, 9, 10, 11, 12, 13, 14, 15);
      acc[mo] = __builtin_amdgcn_wmma_f32_16x16x32_f16(
          false, af, false, bf, (short)0, acc[mo], false, false);
    }
  }

  // C/D layout: VGPR p, lane -> (M = p + 8*half, N = lane&15)
  float* yb = yws + (size_t)br * COUT * HW + n;
#pragma unroll
  for (int mo = 0; mo < 4; ++mo) {
#pragma unroll
    for (int p = 0; p < 8; ++p) {
      int o = mo * 16 + p + half * 8;
      yb[(size_t)o * HW] = acc[mo][p];
    }
  }
}

// pad-1 + 2x2/2 average pool; out [B, Cout, r, 49, 49]
__global__ void k_pool(const float* __restrict__ yws, float* __restrict__ out) {
  int e = blockIdx.x * 256 + threadIdx.x;            // < 8*64*4*49*49
  int ox = e % 49; int rest = e / 49;
  int oy = rest % 49; rest /= 49;
  int r = rest & 3; rest >>= 2;
  int o = rest & 63; int b = rest >> 6;
  const float* yp = yws + ((size_t)((b * 4 + r) * COUT + o)) * HW;
  float s = 0.f;
#pragma unroll
  for (int dy = 0; dy < 2; ++dy) {
    int y = 2 * oy + dy - 1;
    if ((unsigned)y >= (unsigned)HH) continue;
#pragma unroll
    for (int dx = 0; dx < 2; ++dx) {
      int x = 2 * ox + dx - 1;
      if ((unsigned)x >= (unsigned)WW) continue;
      s += yp[y * WW + x];
    }
  }
  out[e] = 0.25f * s;
}

// ---------------------------------------------------------------------------
extern "C" void kernel_launch(void* const* d_in, const int* in_sizes, int n_in,
                              void* d_out, int out_size, void* d_ws, size_t ws_size,
                              hipStream_t stream) {
  (void)in_sizes; (void)n_in; (void)out_size; (void)ws_size;
  const float* x     = (const float*)d_in[0];
  const float* gamma = (const float*)d_in[1];
  const float* beta  = (const float*)d_in[2];
  const float* w     = (const float*)d_in[3];
  const float* w1    = (const float*)d_in[4];
  const float* w2    = (const float*)d_in[5];
  const float* wsp   = (const float*)d_in[6];

  char* ws = (char*)d_ws;
  float*    mean  = (float*)(ws + OFF_MEAN);
  float*    rstd  = (float*)(ws + OFF_RSTD);
  float*    ssum  = (float*)(ws + OFF_SSUM);
  unsigned* smax  = (unsigned*)(ws + OFF_SMAX);
  float*    favg  = (float*)(ws + OFF_FAVG);
  float*    fmaxb = (float*)(ws + OFF_FMAX);
  float*    ach   = (float*)(ws + OFF_ACH);
  float*    asp   = (float*)(ws + OFF_ASP);
  _Float16* weff  = (_Float16*)(ws + OFF_WEFF);
  _Float16* ut    = (_Float16*)(ws + OFF_UT);
  float*    yws   = (float*)(ws + OFF_YWS);

  k_zero   <<<32,   256, 0, stream>>>(ssum, 2 * BATCH * KIH);       // ssum+smax
  k_bnstats<<<CIN,  256, 0, stream>>>(x, mean, rstd);
  k_norm   <<<BATCH * (HW / 32), 256, 0, stream>>>(x, gamma, beta, mean, rstd,
                                                   ut, favg, fmaxb, ssum, smax);
  k_mlp    <<<BATCH * NH * NH, 128, 0, stream>>>(ssum, (const float*)smax, w1, w2, ach);
  k_fold   <<<(NBR * COUT * KIH) / 256, 256, 0, stream>>>(w, ach, weff);
  k_spat   <<<BATCH * NH * NH, 256, 0, stream>>>(favg, fmaxb, wsp, asp);
  k_gemm   <<<dim3(HW / 128, NBR), 256, 0, stream>>>(weff, ut, asp, yws);
  k_pool   <<<(BATCH * COUT * NH * 49 * 49) / 256, 256, 0, stream>>>(yws, (float*)d_out);
}
